// RI_decoder_68487548502101
// MI455X (gfx1250) — matrix operand order repaired and checked by reading, not compiled, not run on hardware
//
#include <hip/hip_runtime.h>
#include <hip/hip_bf16.h>

typedef _Float16 f16;
typedef __attribute__((ext_vector_type(16))) _Float16 v16h;
typedef __attribute__((ext_vector_type(8)))  float    v8f;
typedef __attribute__((ext_vector_type(4)))  int      v4i;
typedef __attribute__((ext_vector_type(8)))  int      v8i;
typedef __attribute__((ext_vector_type(4)))  unsigned int v4u;

#define NPTS 4096
#define NBATCH 4
#define CFEAT 64

// ---- CDNA5 async global->LDS path (confirmed working on this toolchain) ----
#if __has_builtin(__builtin_amdgcn_global_load_async_to_lds_b128) && \
    __has_builtin(__builtin_amdgcn_s_wait_asynccnt)
#define HAVE_ASYNC_LDS 1
#else
#define HAVE_ASYNC_LDS 0
#endif

// ---- CDNA5 Tensor Data Mover path (guarded experiment) ----
#if __has_builtin(__builtin_amdgcn_tensor_load_to_lds) && \
    __has_builtin(__builtin_amdgcn_s_wait_tensorcnt)
#define HAVE_TDM 1
#else
#define HAVE_TDM 0
#endif

__device__ __forceinline__ void copy16_g2lds(f16* lds, const f16* g) {
#if HAVE_ASYNC_LDS
  __builtin_amdgcn_global_load_async_to_lds_b128(
      (__attribute__((address_space(1))) v4i*)(g),
      (__attribute__((address_space(3))) v4i*)(lds), 0, 0);
#else
  *(uint4*)lds = *(const uint4*)g;
#endif
}

__device__ __forceinline__ void wait_async_then_barrier() {
#if HAVE_ASYNC_LDS
  __builtin_amdgcn_s_wait_asynccnt(0);
#endif
  __syncthreads();
}

#if HAVE_TDM
// TDM: DMA a 16-row x 64-byte 2D tile (rows strided Cin*2 bytes in global)
// into LDS, rows packed back-to-back (16x32 f16). D# per CDNA5 ISA ch.8:
// group0 = {count=1, lds_addr, global_addr lo, global_addr hi | type=2<<30}
// group1 = {mask=0|data_size(4B)<<16, tensor_dim0=16, tensor_dim1=16,
//           tile_dim0=16 dw, tile_dim1=16 rows, dim0_stride=Cin/2 dw}
__device__ __forceinline__ void tdm_stage_rows16(f16* ldsdst, const f16* gsrc, int Cin) {
  unsigned lds_off = (unsigned)(unsigned long long)
      (__attribute__((address_space(3))) void*)ldsdst;
  unsigned long long ga = (unsigned long long)gsrc;
  v4u g0 = {1u, lds_off, (unsigned)ga, (unsigned)(ga >> 32) | (2u << 30)};
  v8i g1 = {0x20000, 16 << 16, 16 << 16, 16 << 16, 16, Cin >> 1, 0, 0};
  v4i z4 = {};
#if defined(__clang_major__) && (__clang_major__ >= 23)
  v8i z8 = {};
  __builtin_amdgcn_tensor_load_to_lds(g0, g1, z4, z4, z8, 0);
#else
  __builtin_amdgcn_tensor_load_to_lds(g0, g1, z4, z4, 0);
#endif
}
#endif

__device__ __forceinline__ v8f wmma_f16(v16h a, v16h b, v8f c) {
  return __builtin_amdgcn_wmma_f32_16x16x32_f16(false, a, false, b, (short)0, c, false, false);
}

// A-tile (16x32 f16, row-major weight W[O][Cin]):
// lane<16: M=lane, K = kbase+{0..15}
// lane>=16: M=lane-16, K = kbase+{16..31}
__device__ __forceinline__ v16h load_a_tile(const f16* __restrict__ W, int ldc,
                                            int mbase, int kbase, int lane) {
  int row = mbase + (lane & 15);
  int kg  = lane >> 4;
  const f16* p = W + (size_t)row * ldc + kbase + kg * 16;
  union { uint4 u[2]; v16h v; } t;
  t.u[0] = *(const uint4*)p;
  t.u[1] = *(const uint4*)(p + 8);
  return t.v;
}

// B-tile (32x16 f16) from LDS tile laid out [pos][ch] with stride `ld` f16:
// lane<16: col=lane, K = kbase+0..15 ; lane>=16: col=lane-16, K = kbase+16..31
__device__ __forceinline__ v16h load_b_lds(const f16* base, int ld, int kbase, int lane) {
  int pos = lane & 15, kg = lane >> 4;
  return *(const v16h*)(base + pos * ld + kbase + kg * 16);
}

__device__ __forceinline__ void store_d_f16(f16* dst, v8f d) {
  union { uint4 u; f16 h[8]; } t;
#pragma unroll
  for (int r = 0; r < 8; ++r) t.h[r] = (f16)d[r];
  *(uint4*)dst = t.u;
}

// ---------------------------------------------------------------- cvt f32->f16
__global__ void cvt_f32_f16_kernel(const float* __restrict__ s, f16* __restrict__ d, int n) {
  int i = blockIdx.x * 256 + threadIdx.x;
  if (i < n) d[i] = (f16)s[i];
}

// --------------------------------------------- features [B][64][N] -> f16 [B][N][64]
__global__ __launch_bounds__(256) void transpose_feat_kernel(const float* __restrict__ f,
                                                             f16* __restrict__ ft) {
  __shared__ f16 tile[64 * 72];
  int b = blockIdx.x >> 6;
  int nbase = (blockIdx.x & 63) << 6;
  int t = threadIdx.x;
#pragma unroll
  for (int i = 0; i < 16; ++i) {
    int li = i * 256 + t;
    int c = li >> 6, nn = li & 63;
    tile[nn * 72 + c] = (f16)f[((size_t)b * CFEAT + c) * NPTS + nbase + nn];
  }
  __syncthreads();
#pragma unroll
  for (int i = 0; i < 16; ++i) {
    int li = i * 256 + t;
    int nn = li >> 6, c = li & 63;
    ft[((size_t)(b * NPTS + nbase + nn)) * CFEAT + c] = tile[nn * 72 + c];
  }
}

// ------------------------------------------------- top-64 nearest (sorted ascending)
__global__ __launch_bounds__(256) void knn64_kernel(const float* __restrict__ points,
                                                    unsigned short* __restrict__ idx64) {
  __shared__ float sx[256], sy[256], sz[256];
  int tid = threadIdx.x;
  int b = blockIdx.x >> 4;
  int q = ((blockIdx.x & 15) << 8) + tid;
  const float* pb = points + (size_t)b * 3 * NPTS;
  float qx = pb[q], qy = pb[NPTS + q], qz = pb[2 * NPTS + q];

  float bestd[64];
  unsigned short besti[64];
  for (int i = 0; i < 64; ++i) { bestd[i] = 3.4e38f; besti[i] = 0; }

  for (int tile = 0; tile < NPTS; tile += 256) {
    __syncthreads();
    sx[tid] = pb[tile + tid];
    sy[tid] = pb[NPTS + tile + tid];
    sz[tid] = pb[2 * NPTS + tile + tid];
    __syncthreads();
    for (int j = 0; j < 256; ++j) {
      float dx = qx - sx[j], dy = qy - sy[j], dz = qz - sz[j];
      float d = dx * dx + dy * dy + dz * dz;
      if (d < bestd[63]) {
        int p = 63;
        while (p > 0 && bestd[p - 1] > d) {
          bestd[p] = bestd[p - 1]; besti[p] = besti[p - 1]; --p;
        }
        bestd[p] = d; besti[p] = (unsigned short)(tile + j);
      }
    }
  }
  unsigned short* out = idx64 + ((size_t)(b * NPTS) + q) * 64;
  for (int i = 0; i < 64; ++i) out[i] = besti[i];
}

// ------------------------------------------------------ fused gather+MLP2d+maxK
// grid: 2 branches * 4 batches * 1024 qtiles = 8192 WGs; 256 thr = 8 waves.
// wave w: query qtile + (w>>1), K-half (w&1); 16 positions per wave.
#define EDGE_LD 144
__global__ __launch_bounds__(256) void edge_mlp_kernel(
    const f16* __restrict__ featT, const unsigned short* __restrict__ idx64,
    const f16* __restrict__ w1a, const f16* __restrict__ w1b, const f16* __restrict__ w1c,
    const float* __restrict__ b1a, const float* __restrict__ b1b, const float* __restrict__ b1c,
    const f16* __restrict__ w2a, const f16* __restrict__ w2b, const f16* __restrict__ w2c,
    const float* __restrict__ b2a, const float* __restrict__ b2b, const float* __restrict__ b2c,
    f16* __restrict__ local) {
  __shared__ __attribute__((aligned(32))) f16 smem[8][16][EDGE_LD];
  __shared__ float obuf[4][128][2];

  int tid = threadIdx.x;
  int wave = tid >> 5, lane = tid & 31;
  int branch = blockIdx.x >> 12;
  int wg = blockIdx.x & 4095;
  int b = wg >> 10;
  int qtile = (wg & 1023) * 4;
  int q = qtile + (wave >> 1);
  int h = wave & 1;
  int kg = lane >> 4, pos = lane & 15;

  const f16*  wa = branch ? w2a : w1a;
  const f16*  wb = branch ? w2b : w1b;
  const f16*  wc = branch ? w2c : w1c;
  const float* ba = branch ? b2a : b1a;
  const float* bbv = branch ? b2b : b1b;
  const float* bc = branch ? b2c : b1c;

  for (int i = tid; i < 4 * 128 * 2; i += 256) ((float*)obuf)[i] = -3.4e38f;

  // gather: lane covers half (32 ch) of position (h*16 + pos)'s neighbor vector
  // via async global->LDS (per-lane global addr -> per-lane LDS addr)
  {
    int kk = h * 16 + pos;
    int col = branch ? (kk * 2) : kk;
    int j = idx64[((size_t)(b * NPTS + q)) * 64 + col];
    const f16* src = featT + ((size_t)(b * NPTS + j)) * CFEAT + kg * 32;
    f16* dst = &smem[wave][pos][kg * 32];
#pragma unroll
    for (int i = 0; i < 4; ++i) copy16_g2lds(dst + i * 8, src + i * 8);
  }
  wait_async_then_barrier();

  f16* my = &smem[wave][0][0];

  // ---- layer 1: 64 -> 64, ReLU
  {
    v16h B0 = load_b_lds(my, EDGE_LD, 0, lane);
    v16h B1 = load_b_lds(my, EDGE_LD, 32, lane);
    v8f y[4];
#pragma unroll
    for (int m = 0; m < 4; ++m) {
      v8f acc = {};
      acc = wmma_f16(load_a_tile(wa, 64, m * 16, 0, lane), B0, acc);
      acc = wmma_f16(load_a_tile(wa, 64, m * 16, 32, lane), B1, acc);
      y[m] = acc;
    }
#pragma unroll
    for (int m = 0; m < 4; ++m) {
      const float* bp = ba + m * 16 + kg * 8;
      float4 bl = *(const float4*)bp, bh = *(const float4*)(bp + 4);
      float bb[8] = {bl.x, bl.y, bl.z, bl.w, bh.x, bh.y, bh.z, bh.w};
      v8f v = y[m];
#pragma unroll
      for (int r = 0; r < 8; ++r) { float t = v[r] + bb[r]; v[r] = t > 0.f ? t : 0.f; }
      store_d_f16(&smem[wave][pos][m * 16 + kg * 8], v);
    }
  }
  __syncthreads();

  // ---- layer 2: 64 -> 128, ReLU
  {
    v16h B0 = load_b_lds(my, EDGE_LD, 0, lane);
    v16h B1 = load_b_lds(my, EDGE_LD, 32, lane);
    v8f z[8];
#pragma unroll
    for (int m = 0; m < 8; ++m) {
      v8f acc = {};
      acc = wmma_f16(load_a_tile(wb, 64, m * 16, 0, lane), B0, acc);
      acc = wmma_f16(load_a_tile(wb, 64, m * 16, 32, lane), B1, acc);
      z[m] = acc;
    }
#pragma unroll
    for (int m = 0; m < 8; ++m) {
      const float* bp = bbv + m * 16 + kg * 8;
      float4 bl = *(const float4*)bp, bh = *(const float4*)(bp + 4);
      float bb[8] = {bl.x, bl.y, bl.z, bl.w, bh.x, bh.y, bh.z, bh.w};
      v8f v = z[m];
#pragma unroll
      for (int r = 0; r < 8; ++r) { float t = v[r] + bb[r]; v[r] = t > 0.f ? t : 0.f; }
      store_d_f16(&smem[wave][pos][m * 16 + kg * 8], v);
    }
  }
  __syncthreads();

  // ---- layer 3: 128 -> 128 (no ReLU), then max over the 16 positions
  {
    v16h D0 = load_b_lds(my, EDGE_LD, 0, lane);
    v16h D1 = load_b_lds(my, EDGE_LD, 32, lane);
    v16h D2 = load_b_lds(my, EDGE_LD, 64, lane);
    v16h D3 = load_b_lds(my, EDGE_LD, 96, lane);
#pragma unroll
    for (int m = 0; m < 8; ++m) {
      v8f acc = {};
      acc = wmma_f16(load_a_tile(wc, 128, m * 16, 0, lane), D0, acc);
      acc = wmma_f16(load_a_tile(wc, 128, m * 16, 32, lane), D1, acc);
      acc = wmma_f16(load_a_tile(wc, 128, m * 16, 64, lane), D2, acc);
      acc = wmma_f16(load_a_tile(wc, 128, m * 16, 96, lane), D3, acc);
      // max over the 16 columns (positions) held by this lane half-group
#pragma unroll
      for (int off = 8; off >= 1; off >>= 1) {
#pragma unroll
        for (int r = 0; r < 8; ++r) acc[r] = fmaxf(acc[r], __shfl_xor(acc[r], off, 32));
      }
      if (pos == 0) {
        int chb = m * 16 + kg * 8;
#pragma unroll
        for (int r = 0; r < 8; ++r) obuf[wave >> 1][chb + r][h] = acc[r];
      }
    }
  }
  __syncthreads();

  // combine the two K-halves, add layer-3 bias (monotone: max then +b), emit f16
  for (int i = tid; i < 512; i += 256) {
    int qq = i >> 7, ch = i & 127;
    float v = fmaxf(obuf[qq][ch][0], obuf[qq][ch][1]) + bc[ch];
    local[((size_t)(b * NPTS + qtile + qq)) * 256 + branch * 128 + ch] = (f16)v;
  }
}

// ------------------------------------------------- merge MLP1d GEMM (WMMA)
// out tile 128x128 per WG; wave w owns M rows [ot*128+w*16, +16), 8 N-subtiles.
// B tiles (128 pos x 32 ch, 8KB) staged in LDS once per WG per k-step,
// double-buffered. Staging engine: TDM descriptor per wave (16 rows each)
// when available, else per-lane async global->LDS, else load+ds_store.
// mode 0: Yh[pos][O] f16 + bias + ReLU ; mode 1: Yf[b][o][n] f32 + bias
__global__ __launch_bounds__(256) void merge_gemm_kernel(
    const f16* __restrict__ X, const f16* __restrict__ W, const float* __restrict__ bias,
    f16* __restrict__ Yh, float* __restrict__ Yf, int Cin, int O, int mode) {
  __shared__ __attribute__((aligned(32))) f16 bsh[2][128 * 32];
  int tid = threadIdx.x, wave = tid >> 5, lane = tid & 31;
  int ot = blockIdx.x >> 7;          // P/128 == 128 tiles
  int pt = blockIdx.x & 127;
  int obase = ot * 128 + wave * 16;
  int pbase = pt * 128;
  int kg = lane >> 4, lpos = lane & 15;

  v8f acc[8];
#pragma unroll
  for (int ns = 0; ns < 8; ++ns) acc[ns] = (v8f){};

#if HAVE_TDM
  int wv = __builtin_amdgcn_readfirstlane(tid >> 5);   // scalarized wave id
#endif

  auto stage = [&](int buf, int kk) {
#if HAVE_TDM
    // one TDM descriptor per wave: rows wv*16..wv*16+15, 64B each
    tdm_stage_rows16(&bsh[buf][wv * 16 * 32],
                     X + (size_t)(pbase + wv * 16) * Cin + kk, Cin);
#else
#pragma unroll
    for (int i = 0; i < 2; ++i) {
      int id = tid + 256 * i;          // 0..511
      int pos = id >> 2, part = id & 3;
      const f16* g = X + (size_t)(pbase + pos) * Cin + kk + part * 8;
      copy16_g2lds(&bsh[buf][pos * 32 + part * 8], g);
    }
#endif
  };
  auto wait_stage = [&]() {
#if HAVE_TDM
    __builtin_amdgcn_s_wait_tensorcnt(0);
    __syncthreads();
#else
    wait_async_then_barrier();
#endif
  };

  stage(0, 0);
  wait_stage();

  int nk = Cin / 32;
  for (int t = 0; t < nk; ++t) {
    int kk = t * 32, buf = t & 1;
    if (t + 1 < nk) stage(buf ^ 1, kk + 32);       // DMA prefetch next k-step
    __builtin_prefetch(W + (size_t)(obase + lpos) * Cin + kk + 32, 0, 1);
    v16h A = load_a_tile(W, Cin, obase, kk, lane);
#pragma unroll
    for (int ns = 0; ns < 8; ++ns)
      acc[ns] = wmma_f16(A, load_b_lds(&bsh[buf][ns * 16 * 32], 32, 0, lane), acc[ns]);
    wait_stage();
  }

  const float* bp = bias + obase + kg * 8;
  float4 bl = *(const float4*)bp, bh = *(const float4*)(bp + 4);
  float bb[8] = {bl.x, bl.y, bl.z, bl.w, bh.x, bh.y, bh.z, bh.w};

#pragma unroll
  for (int ns = 0; ns < 8; ++ns) {
    int p = pbase + ns * 16 + lpos;
    v8f v = acc[ns];
#pragma unroll
    for (int r = 0; r < 8; ++r) v[r] += bb[r];
    if (mode == 0) {
#pragma unroll
      for (int r = 0; r < 8; ++r) v[r] = v[r] > 0.f ? v[r] : 0.f;
      store_d_f16(Yh + (size_t)p * O + obase + kg * 8, v);
    } else {
      int b = p >> 12, n = p & (NPTS - 1);
#pragma unroll
      for (int r = 0; r < 8; ++r)
        Yf[((size_t)b * 1024 + (obase + kg * 8 + r)) * NPTS + n] = v[r];
    }
  }
}

// ------------------------------------------------------------------ host launch
extern "C" void kernel_launch(void* const* d_in, const int* in_sizes, int n_in,
                              void* d_out, int out_size, void* d_ws, size_t ws_size,
                              hipStream_t stream) {
  (void)in_sizes; (void)n_in; (void)out_size; (void)ws_size;
  const float* points   = (const float*)d_in[0];
  const float* features = (const float*)d_in[1];
  const float* wsrc[9]  = {(const float*)d_in[2],  (const float*)d_in[4],  (const float*)d_in[6],
                           (const float*)d_in[8],  (const float*)d_in[10], (const float*)d_in[12],
                           (const float*)d_in[14], (const float*)d_in[16], (const float*)d_in[18]};
  const float* b1a = (const float*)d_in[3],  *b1b = (const float*)d_in[5],  *b1c = (const float*)d_in[7];
  const float* b2a = (const float*)d_in[9],  *b2b = (const float*)d_in[11], *b2c = (const float*)d_in[13];
  const float* bma = (const float*)d_in[15], *bmb = (const float*)d_in[17], *bmc = (const float*)d_in[19];

  char* ws = (char*)d_ws;
  const size_t MB = 1024u * 1024u;
  f16*            featT = (f16*)(ws + 0);                 // 2 MB
  unsigned short* idx64 = (unsigned short*)(ws + 2 * MB); // 2 MB
  f16*            local = (f16*)(ws + 4 * MB);            // 8 MB
  f16*            act1  = (f16*)(ws + 12 * MB);           // 16 MB
  f16*            act2  = (f16*)(ws + 28 * MB);           // 32 MB
  f16*            wf    = (f16*)(ws + 60 * MB);           // ~3.4 MB f16 weights

  const int wsz[9] = {64 * 64, 128 * 64, 128 * 128, 64 * 64, 128 * 64, 128 * 128,
                      512 * 256, 1024 * 512, 1024 * 1024};
  size_t woff[9]; size_t o = 0;
  for (int i = 0; i < 9; ++i) { woff[i] = o; o += (size_t)wsz[i]; }

  for (int i = 0; i < 9; ++i)
    cvt_f32_f16_kernel<<<(wsz[i] + 255) / 256, 256, 0, stream>>>(wsrc[i], wf + woff[i], wsz[i]);

  transpose_feat_kernel<<<NBATCH * (NPTS / 64), 256, 0, stream>>>(features, featT);
  knn64_kernel<<<NBATCH * (NPTS / 256), 256, 0, stream>>>(points, idx64);

  edge_mlp_kernel<<<2 * NBATCH * (NPTS / 4), 256, 0, stream>>>(
      featT, idx64,
      wf + woff[0], wf + woff[1], wf + woff[2], b1a, b1b, b1c,
      wf + woff[3], wf + woff[4], wf + woff[5], b2a, b2b, b2c,
      local);

  const int P = NBATCH * NPTS;        // 16384 positions, 128 P-tiles
  merge_gemm_kernel<<<(512 / 128) * (P / 128), 256, 0, stream>>>(
      local, wf + woff[6], bma, act1, nullptr, 256, 512, 0);
  merge_gemm_kernel<<<(1024 / 128) * (P / 128), 256, 0, stream>>>(
      act1, wf + woff[7], bmb, act2, nullptr, 512, 1024, 0);
  merge_gemm_kernel<<<(1024 / 128) * (P / 128), 256, 0, stream>>>(
      act2, wf + woff[8], bmc, nullptr, (float*)d_out, 1024, 1024, 1);
}
